// PositionOpener_2731599200467
// MI455X (gfx1250) — matrix operand order repaired and checked by reading, not compiled, not run on hardware
//
#include <hip/hip_runtime.h>

#define HIDDEN   512
#define BATCH    128
#define NCH      16
#define CHUNK    64
#define TSTEPS   1024      // NCH * CHUNK
#define BT_TILE  16        // batch rows per workgroup
#define NBT      8         // batch tiles
#define NSL      8         // hidden-slice workgroups per batch tile
#define SL_H     64        // hidden units per workgroup (4 waves * 16)
#define THREADS  128       // 4 waves of 32
#define PROBS_N  (BATCH * TSTEPS)                  // 131072
#define HX_N     (NCH * 2 * BATCH * HIDDEN)        // 2097152

typedef __attribute__((ext_vector_type(16))) _Float16 v16h;
typedef __attribute__((ext_vector_type(8)))  float    v8f;
typedef __attribute__((ext_vector_type(4)))  _Float16 h4;
typedef __attribute__((ext_vector_type(4)))  unsigned u32x4;
typedef __attribute__((ext_vector_type(4)))  float    f32x4;
typedef __attribute__((ext_vector_type(2)))  float    f32x2;

union V16H { u32x4 u[2]; v16h v; };

__device__ __forceinline__ float fast_tanh(float x) {
#if __has_builtin(__builtin_amdgcn_tanhf)
    return __builtin_amdgcn_tanhf(x);
#else
    float e = __expf(2.0f * x);
    return 1.0f - 2.0f / (e + 1.0f);
#endif
}
__device__ __forceinline__ float fast_sig(float x) {
    return 0.5f * fast_tanh(0.5f * x) + 0.5f;
}

struct PODf2 { float x, y; };

// ---------------------------------------------------------------------------
// Init: zero r-accumulation region of d_out, the h exchange buffers, counters.
// ---------------------------------------------------------------------------
__global__ void init_kernel(float* out, _Float16* hbuf, unsigned* counters) {
    int i = blockIdx.x * blockDim.x + threadIdx.x;
    int stride = gridDim.x * blockDim.x;
    for (int j = i; j < PROBS_N; j += stride) out[j] = 0.0f;
    // hbuf: 2 * NBT * 16 * HIDDEN halves = 65536 floats worth
    float* hb = (float*)hbuf;
    for (int j = i; j < (2 * NBT * BT_TILE * HIDDEN) / 2; j += stride) hb[j] = 0.0f;
    if (i < NBT) counters[i] = 0u;
}

// ---------------------------------------------------------------------------
// Persistent LSTM kernel. Grid: NBT*NSL workgroups of 128 threads.
// WG (bt, sl) owns batch rows [bt*16, bt*16+16) and hidden slice
// [sl*64, sl*64+64); wave v owns 16 hidden units within the slice.
// W_hh slice lives in LDS (f16, XOR-swizzled); h is exchanged per step via a
// double-buffered f16 buffer in global memory + an atomic-counter barrier.
// ---------------------------------------------------------------------------
__global__ __launch_bounds__(THREADS, 1)
void lstm_persistent(const int* __restrict__ inds, const float* __restrict__ p,
                     const float* __restrict__ W_ih, const float* __restrict__ W_hh,
                     const float* __restrict__ b_ih, const float* __restrict__ b_hh,
                     const float* __restrict__ W_lin,
                     float* __restrict__ out,
                     _Float16* __restrict__ hbuf,   // [2][NBT][16][HIDDEN] f16
                     unsigned* __restrict__ counters) {
    __shared__ _Float16 Wl[4 * SL_H * HIDDEN];     // 256 KB: [gate][row-in-slice][K] swizzled
    __shared__ float    cst[4 * SL_H * 4];         // 4 KB: per gate-row (wi0, wi1, bias, 0)
    __shared__ PODf2    xc[CHUNK * BT_TILE];       // 8 KB: per-chunk inputs x[t][M]
    __shared__ int      sh_inds[BT_TILE];

    const int bt   = blockIdx.x / NSL;
    const int sl   = blockIdx.x % NSL;
    const int tid  = threadIdx.x;
    const int wave = tid >> 5;
    const int lane = tid & 31;
    const int ln   = lane & 15;      // N within tile
    const int hi   = lane >> 4;      // half-wave select

    // ---- one-time: convert W_hh slice f32 -> f16 into swizzled LDS ----
    for (int e4 = tid; e4 < (4 * SL_H * HIDDEN) / 4; e4 += THREADS) {
        int r = e4 >> 7;                 // row 0..255 (= g*64 + j)
        int k = (e4 & 127) << 2;         // K, multiple of 4
        int g = r >> 6, j = r & 63;
        int gr = g * HIDDEN + sl * SL_H + j;     // gate row in [0,2048) per gate block
        int grow = g * 512 + sl * SL_H + j;      // global W_hh row
        (void)gr;
        f32x4 w = *(const f32x4*)(W_hh + (size_t)grow * HIDDEN + k);
        h4 hv;
        hv.x = (_Float16)w.x; hv.y = (_Float16)w.y;
        hv.z = (_Float16)w.z; hv.w = (_Float16)w.w;
        // 8-half-chunk XOR swizzle for bank-conflict-free B fragment reads
        int ksw = (((k >> 3) ^ (r & 15)) << 3) | (k & 7);
        *(h4*)&Wl[r * HIDDEN + ksw] = hv;
    }
    for (int r = tid; r < 4 * SL_H; r += THREADS) {
        int g = r >> 6, j = r & 63;
        int grow = g * 512 + sl * SL_H + j;
        cst[r * 4 + 0] = W_ih[grow * 2 + 0];
        cst[r * 4 + 1] = W_ih[grow * 2 + 1];
        cst[r * 4 + 2] = b_ih[grow] + b_hh[grow];
        cst[r * 4 + 3] = 0.0f;
    }
    if (tid < BT_TILE) sh_inds[tid] = inds[bt * BT_TILE + tid];
    __syncthreads();

    const int n_local = wave * 16 + ln;            // 0..63 within slice
    const int n_glob  = sl * SL_H + n_local;       // hidden index 0..511
    const float w1 = W_lin[HIDDEN + n_glob];       // W_lin[1][n]

    // cell state lives entirely in registers (accumulator layout tile)
    float c[8];
#pragma unroll
    for (int vv = 0; vv < 8; ++vv) c[vv] = 0.0f;

    for (int t = 0; t < TSTEPS; ++t) {
        const int ch = t >> 6, tin = t & 63;

        if (tin == 0) {
            // load this chunk's inputs: x[t][M] = p[ind+ch*64+t] - p[ind+ch*64]
            for (int e = tid; e < CHUNK * BT_TILE; e += THREADS) {
                int ts = e >> 4, M = e & 15;
                int base = sh_inds[M] + ch * CHUNK;
                f32x2 v  = ((const f32x2*)p)[base + ts];
                f32x2 v0 = ((const f32x2*)p)[base];
                xc[ts * BT_TILE + M].x = v.x - v0.x;
                xc[ts * BT_TILE + M].y = v.y - v0.y;
            }
            __syncthreads();
        }

        // x for my 8 batch rows (M = vv + 8*hi)
        float xa[8], xb[8];
        {
            const PODf2* xrow = &xc[tin * BT_TILE + 8 * hi];
#pragma unroll
            for (int vv = 0; vv < 8; ++vv) { xa[vv] = xrow[vv].x; xb[vv] = xrow[vv].y; }
        }

        // init accumulators with bias + x @ W_ih^T
        v8f acc[4];
#pragma unroll
        for (int g = 0; g < 4; ++g) {
            const float* cc = &cst[(g * SL_H + n_local) * 4];
            float wi0 = cc[0], wi1 = cc[1], bb = cc[2];
#pragma unroll
            for (int vv = 0; vv < 8; ++vv)
                acc[g][vv] = bb + xa[vv] * wi0 + xb[vv] * wi1;
        }

        // h @ W_hh^T via WMMA: A = h tile (16x32 f16 from global exchange buf),
        // B = W slice tile (32x16 f16 from LDS), accumulate f32.
        const _Float16* hb = hbuf + ((size_t)(t & 1) * NBT + bt) * (BT_TILE * HIDDEN);
        const _Float16* arow = hb + (size_t)ln * HIDDEN + hi * 8;
        for (int kt = 0; kt < 16; ++kt) {
            V16H A;
            const _Float16* ap = arow + kt * 32;
            A.u[0] = *(const u32x4*)(ap);
            A.u[1] = *(const u32x4*)(ap + 16);
            int q8 = kt * 4 + hi;
#pragma unroll
            for (int g = 0; g < 4; ++g) {
                const _Float16* wrow = &Wl[(g * SL_H + n_local) * HIDDEN];
                V16H B;
                B.u[0] = *(const u32x4*)(wrow + (((q8)     ^ ln) << 3));
                B.u[1] = *(const u32x4*)(wrow + (((q8 + 2) ^ ln) << 3));
                acc[g] = __builtin_amdgcn_wmma_f32_16x16x32_f16(
                    false, A.v, false, B.v, (short)0, acc[g], false, false);
            }
        }

        // LSTM cell update (i, f, g, o) + fused W_lin[1] partial dot
        float hn[8], pr[8];
#pragma unroll
        for (int vv = 0; vv < 8; ++vv) {
            float ig = fast_sig(acc[0][vv]);
            float fg = fast_sig(acc[1][vv]);
            float gg = fast_tanh(acc[2][vv]);
            float og = fast_sig(acc[3][vv]);
            float cv = fg * c[vv] + ig * gg;
            c[vv] = cv;
            float hv = og * fast_tanh(cv);
            hn[vv] = hv;
            pr[vv] = hv * w1;
        }

        // publish new h (f16) into the other buffer
        _Float16* hw = hbuf + ((size_t)((t + 1) & 1) * NBT + bt) * (BT_TILE * HIDDEN) + n_glob;
#pragma unroll
        for (int vv = 0; vv < 8; ++vv)
            hw[(size_t)(vv + 8 * hi) * HIDDEN] = (_Float16)hn[vv];

        // reduce r-projection partials over the 16 lanes of each half-wave
#pragma unroll
        for (int m = 1; m < 16; m <<= 1) {
#pragma unroll
            for (int vv = 0; vv < 8; ++vv)
                pr[vv] += __shfl_xor(pr[vv], m, 32);
        }
        if (ln == 0) {
#pragma unroll
            for (int vv = 0; vv < 8; ++vv)
                atomicAdd(out + (size_t)(bt * 16 + vv + 8 * hi) * TSTEPS + t, pr[vv]);
        }

        // chunk boundary: dump (h_end, c_end) -> open_hx [ch][0/1][B][H]
        if (tin == CHUNK - 1) {
            float* hx = out + PROBS_N + (size_t)(ch * 2 + 0) * BATCH * HIDDEN;
            float* cx = out + PROBS_N + (size_t)(ch * 2 + 1) * BATCH * HIDDEN;
#pragma unroll
            for (int vv = 0; vv < 8; ++vv) {
                size_t off = (size_t)(bt * 16 + vv + 8 * hi) * HIDDEN + n_glob;
                hx[off] = hn[vv];
                cx[off] = c[vv];
            }
        }

        // cross-WG barrier among the NSL slice workgroups of this batch tile
        __threadfence();
        __syncthreads();
        if (tid == 0) {
            __hip_atomic_fetch_add(&counters[bt], 1u, __ATOMIC_RELEASE,
                                   __HIP_MEMORY_SCOPE_AGENT);
            unsigned tgt = (unsigned)(NSL * (t + 1));
            while (__hip_atomic_load(&counters[bt], __ATOMIC_ACQUIRE,
                                     __HIP_MEMORY_SCOPE_AGENT) < tgt) {
                __builtin_amdgcn_s_sleep(1);
            }
        }
        __syncthreads();
    }
}

// ---------------------------------------------------------------------------
// Finalize: r = sigmoid(pre + b_lin[1]); in-place cumprod adjustment and
// survival probability, per the reference.
// ---------------------------------------------------------------------------
__global__ void finalize_kernel(const float* __restrict__ b_lin,
                                float* __restrict__ out) {
    int b = threadIdx.x;
    if (b >= BATCH) return;
    float b1 = b_lin[1];
    float* row = out + (size_t)b * TSTEPS;
    float last_nd = 1.0f, cum = 0.0f;
    for (int ch = 0; ch < NCH; ++ch) {
        float run = 1.0f;   // running prod of (1 - r) within chunk
        for (int tin = 0; tin < CHUNK; ++tin) {
            float pre = row[ch * CHUNK + tin];
            float r = 1.0f / (1.0f + __expf(-(pre + b1)));
            float adj = (tin == 0) ? r : r * last_nd * run;
            row[ch * CHUNK + tin] = adj;
            cum += adj;
            if (tin < CHUNK - 1) run *= (1.0f - r);
        }
        last_nd *= run;     // chunk_prod = cp[:, :, -1]
    }
    out[PROBS_N + HX_N + b] = 1.0f - cum;
}

// ---------------------------------------------------------------------------
extern "C" void kernel_launch(void* const* d_in, const int* in_sizes, int n_in,
                              void* d_out, int out_size, void* d_ws, size_t ws_size,
                              hipStream_t stream) {
    (void)in_sizes; (void)n_in; (void)out_size; (void)ws_size;
    const int*   inds  = (const int*)d_in[0];
    const float* p     = (const float*)d_in[1];
    const float* W_ih  = (const float*)d_in[2];
    const float* W_hh  = (const float*)d_in[3];
    const float* b_ih  = (const float*)d_in[4];
    const float* b_hh  = (const float*)d_in[5];
    const float* W_lin = (const float*)d_in[6];
    const float* b_lin = (const float*)d_in[7];
    float* out = (float*)d_out;

    _Float16* hbuf = (_Float16*)d_ws;                       // 256 KB double buffer
    unsigned* counters = (unsigned*)((char*)d_ws +
        (size_t)2 * NBT * BT_TILE * HIDDEN * sizeof(_Float16));  // offset 262144

    init_kernel<<<64, 256, 0, stream>>>(out, hbuf, counters);
    lstm_persistent<<<NBT * NSL, THREADS, 0, stream>>>(
        inds, p, W_ih, W_hh, b_ih, b_hh, W_lin, out, hbuf, counters);
    finalize_kernel<<<1, 128, 0, stream>>>(b_lin, out);
}
